// TransformerBasedLayer_86852828659816
// MI455X (gfx1250) — compile-verified, matrix operand order checked by reading, not run on hardware
//
#include <hip/hip_runtime.h>
#include <hip/hip_bf16.h>
#include <cmath>

// ---------------------------------------------------------------------------
// TransformerConv (PyG-style) + BatchNorm + LeakyReLU for MI455X (gfx1250).
// N=50000 nodes, E=800000 edges, F=128, H=4 heads, C=32.
// Dense projections use v_wmma_f32_16x16x32_f16 (wave32 WMMA).
// ---------------------------------------------------------------------------

#define NN   50000
#define EE   800000
#define FF   128
#define HH   4
#define CC   32
#define EPSB 1e-5f
#define SLOPE 0.01f
#define INV_SQRT_C 0.17677669529663687f   // 1/sqrt(32)

typedef __attribute__((ext_vector_type(16))) _Float16 v16h;
typedef __attribute__((ext_vector_type(8)))  _Float16 v8h;
typedef __attribute__((ext_vector_type(8)))  float    v8f;

// ------------------------- helpers -----------------------------------------

__device__ __forceinline__ void atomicMaxFloat(float* addr, float val) {
    // float max via int/uint monotonic mapping; init value must be -inf.
    if (val >= 0.0f)
        atomicMax((int*)addr, __float_as_int(val));
    else
        atomicMin((unsigned int*)addr, __float_as_uint(val));
}

// ------------------- K0: pack weights (f32 [k][n] -> f16 [w][n][k]) ---------

__global__ void pack_weights_kernel(const float* __restrict__ Wq,
                                    const float* __restrict__ Wk,
                                    const float* __restrict__ Wv,
                                    const float* __restrict__ Ws,
                                    _Float16* __restrict__ wpack) {
    int idx = blockIdx.x * blockDim.x + threadIdx.x;
    if (idx >= 4 * FF * FF) return;
    int w   = idx >> 14;        // / (128*128)
    int rem = idx & 16383;
    int n   = rem >> 7;         // packed row  = output column
    int k   = rem & 127;        // packed col  = reduction index (contiguous)
    const float* W = (w == 0) ? Wq : (w == 1) ? Wk : (w == 2) ? Wv : Ws;
    wpack[idx] = (_Float16)W[k * FF + n];
}

// ------------------- K1: init scratch ---------------------------------------

__global__ void init_kernel(float* __restrict__ amax, float* __restrict__ denom,
                            float* __restrict__ agg,  float* __restrict__ bnsum,
                            float* __restrict__ bnsq) {
    int i = blockIdx.x * blockDim.x + threadIdx.x;
    if (i < NN * HH) { amax[i] = -INFINITY; denom[i] = 0.0f; }
    if (i < NN * FF) agg[i] = 0.0f;
    if (i < FF)      { bnsum[i] = 0.0f; bnsq[i] = 0.0f; }
}

// ------------------- K2: fused q/k/v/skip GEMM via WMMA ---------------------
// One wave (32 threads) per 16-row x 16-col tile; the A (x) fragment is
// loaded once per k-step and reused for the 4 weight matrices (4 WMMAs).
// K = 128 -> 4 k-steps of 32. A/B fragment addressing follows the CDNA5
// ISA 7.12.2 wave32 16-bit layouts (lane&15 = row/col, lane>>4 selects the
// 8-wide K half; two contiguous 16B chunks per fragment per lane).

__global__ void gemm_qkvs_kernel(const float*    __restrict__ x,
                                 const _Float16* __restrict__ wpack,
                                 const float* __restrict__ bq, const float* __restrict__ bk,
                                 const float* __restrict__ bv, const float* __restrict__ bs,
                                 float* __restrict__ q,  float* __restrict__ k2,
                                 float* __restrict__ v2, float* __restrict__ s2) {
    const int lane  = threadIdx.x;            // blockDim.x == 32
    const int m0    = blockIdx.x * 16;        // node-tile
    const int n0    = blockIdx.y * 16;        // channel-tile (8 tiles of 16)
    const int mrow  = m0 + (lane & 15);
    const int nrow  = n0 + (lane & 15);
    const int khalf = (lane >> 4) * 8;

    v8f acc[4] = {};
    const float* xrow = x + (size_t)mrow * FF;

    #pragma unroll
    for (int kk = 0; kk < FF; kk += 32) {
        // A fragment: row mrow, K = kk+khalf..+7 and kk+16+khalf..+7
        v16h a;
        #pragma unroll
        for (int j = 0; j < 8; ++j) {
            a[j]     = (_Float16)xrow[kk + khalf + j];
            a[8 + j] = (_Float16)xrow[kk + 16 + khalf + j];
        }
        #pragma unroll
        for (int w = 0; w < 4; ++w) {
            const _Float16* wt = wpack + ((size_t)w * FF + nrow) * FF;
            v8h b0 = *(const v8h*)(wt + kk + khalf);
            v8h b1 = *(const v8h*)(wt + kk + 16 + khalf);
            v16h b;
            #pragma unroll
            for (int j = 0; j < 8; ++j) { b[j] = b0[j]; b[8 + j] = b1[j]; }
            acc[w] = __builtin_amdgcn_wmma_f32_16x16x32_f16(
                         false, a, false, b, (short)0, acc[w], false, false);
        }
    }

    const int ncol = n0 + (lane & 15);
    float bias[4] = { bq[ncol], bk[ncol], bv[ncol], bs[ncol] };
    float* outs[4] = { q, k2, v2, s2 };
    #pragma unroll
    for (int w = 0; w < 4; ++w) {
        #pragma unroll
        for (int r = 0; r < 8; ++r) {
            int row = m0 + r + ((lane >> 4) * 8);   // C/D layout: VGPR r -> M=r / M=8+r
            outs[w][(size_t)row * FF + ncol] = acc[w][r] + bias[w];
        }
    }
}

// ------------------- K3: per-edge logits + segment max ----------------------
// One wave per edge; lane = channel within head; coalesced 128B gathers.

__global__ void edge_logits_kernel(const int* __restrict__ ei,
                                   const float* __restrict__ ea,
                                   const float* __restrict__ We,
                                   const float* __restrict__ q,
                                   const float* __restrict__ k,
                                   float* __restrict__ alpha,
                                   float* __restrict__ amax) {
    int eid = blockIdx.x * (blockDim.x >> 5) + (threadIdx.x >> 5);
    if (eid >= EE) return;
    int lane = threadIdx.x & 31;
    int src = ei[eid];
    int dst = ei[EE + eid];
    float ea0 = ea[2 * eid];
    float ea1 = ea[2 * eid + 1];
    const float* qd = q + (size_t)dst * FF;
    const float* ks = k + (size_t)src * FF;
    #pragma unroll
    for (int h = 0; h < HH; ++h) {
        int c = h * CC + lane;
        float emod = ea0 * We[c] + ea1 * We[FF + c];
        float p = qd[c] * (ks[c] + emod);
        #pragma unroll
        for (int off = 16; off > 0; off >>= 1) p += __shfl_xor(p, off);
        if (lane == 0) {
            float a = p * INV_SQRT_C;
            alpha[(size_t)eid * HH + h] = a;
            atomicMaxFloat(&amax[dst * HH + h], a);
        }
    }
}

// ------------------- K4: exp(alpha - max) + segment sum ---------------------

__global__ void edge_expsum_kernel(const int* __restrict__ ei,
                                   float* __restrict__ alpha,
                                   const float* __restrict__ amax,
                                   float* __restrict__ denom) {
    int i = blockIdx.x * blockDim.x + threadIdx.x;
    if (i >= EE * HH) return;
    int eid = i >> 2;
    int h   = i & 3;
    int dst = ei[EE + eid];
    float ex = __expf(alpha[i] - amax[dst * HH + h]);
    alpha[i] = ex;                     // overwrite with exp value
    atomicAdd(&denom[dst * HH + h], ex);
}

// ------------------- K5: weighted messages, scatter-add ---------------------

__global__ void edge_message_kernel(const int* __restrict__ ei,
                                    const float* __restrict__ ea,
                                    const float* __restrict__ We,
                                    const float* __restrict__ v,
                                    const float* __restrict__ alpha,
                                    const float* __restrict__ denom,
                                    float* __restrict__ agg) {
    int eid = blockIdx.x * (blockDim.x >> 5) + (threadIdx.x >> 5);
    if (eid >= EE) return;
    int lane = threadIdx.x & 31;
    int src = ei[eid];
    int dst = ei[EE + eid];
    float ea0 = ea[2 * eid];
    float ea1 = ea[2 * eid + 1];
    const float* vs = v + (size_t)src * FF;
    float* ad = agg + (size_t)dst * FF;
    #pragma unroll
    for (int h = 0; h < HH; ++h) {
        float attn = alpha[(size_t)eid * HH + h] / denom[dst * HH + h];
        int c = h * CC + lane;
        float emod = ea0 * We[c] + ea1 * We[FF + c];
        atomicAdd(&ad[c], (vs[c] + emod) * attn);
    }
}

// ------------------- K6: add skip path + BN batch statistics ----------------
// blockDim = 128 (one thread per channel), each block covers 64 rows,
// partial sums reduced once per block into global accumulators.

__global__ void skip_stats_kernel(const float* __restrict__ skipb,
                                  float* __restrict__ agg,
                                  float* __restrict__ bnsum,
                                  float* __restrict__ bnsq) {
    int c  = threadIdx.x;               // 0..127
    int r0 = blockIdx.x * 64;
    int r1 = r0 + 64; if (r1 > NN) r1 = NN;
    float s = 0.0f, sq = 0.0f;
    for (int r = r0; r < r1; ++r) {
        size_t idx = (size_t)r * FF + c;
        float val = agg[idx] + skipb[idx];
        agg[idx] = val;
        s  += val;
        sq += val * val;
    }
    atomicAdd(&bnsum[c], s);
    atomicAdd(&bnsq[c],  sq);
}

// ------------------- K7: BatchNorm + LeakyReLU ------------------------------

__global__ void finalize_kernel(const float* __restrict__ agg,
                                const float* __restrict__ bnsum,
                                const float* __restrict__ bnsq,
                                const float* __restrict__ gamma,
                                const float* __restrict__ beta,
                                float* __restrict__ out) {
    int i = blockIdx.x * blockDim.x + threadIdx.x;
    if (i >= NN * FF) return;
    int c = i & 127;
    const float invN = 1.0f / (float)NN;
    float mean = bnsum[c] * invN;
    float var  = bnsq[c] * invN - mean * mean;
    float val  = (agg[i] - mean) * rsqrtf(var + EPSB) * gamma[c] + beta[c];
    out[i] = (val > 0.0f) ? val : SLOPE * val;
}

// ------------------------------- launcher -----------------------------------

extern "C" void kernel_launch(void* const* d_in, const int* in_sizes, int n_in,
                              void* d_out, int out_size, void* d_ws, size_t ws_size,
                              hipStream_t stream) {
    const float* x   = (const float*)d_in[0];
    const int*   ei  = (const int*)  d_in[1];   // [2,E] row-major: src then dst
    const float* ea  = (const float*)d_in[2];
    const float* Wq  = (const float*)d_in[3];
    const float* bq  = (const float*)d_in[4];
    const float* Wk  = (const float*)d_in[5];
    const float* bk  = (const float*)d_in[6];
    const float* Wv  = (const float*)d_in[7];
    const float* bv  = (const float*)d_in[8];
    const float* We  = (const float*)d_in[9];
    const float* Wsk = (const float*)d_in[10];
    const float* bsk = (const float*)d_in[11];
    const float* gma = (const float*)d_in[12];
    const float* bta = (const float*)d_in[13];
    float* out = (float*)d_out;

    // ---- workspace layout (all 256B aligned) ----
    char* ws = (char*)d_ws;
    size_t off = 0;
    auto take = [&](size_t bytes) {
        char* p = ws + off;
        off += (bytes + 255) & ~(size_t)255;
        return p;
    };
    _Float16* wpack = (_Float16*)take((size_t)4 * FF * FF * 2);  // 128 KB
    float* q     = (float*)take((size_t)NN * FF * 4);            // 25.6 MB
    float* k2    = (float*)take((size_t)NN * FF * 4);
    float* v2    = (float*)take((size_t)NN * FF * 4);
    float* skipb = (float*)take((size_t)NN * FF * 4);
    float* alpha = (float*)take((size_t)EE * HH * 4);            // 12.8 MB
    float* amax  = (float*)take((size_t)NN * HH * 4);
    float* denom = (float*)take((size_t)NN * HH * 4);
    float* agg   = (float*)take((size_t)NN * FF * 4);
    float* bnsum = (float*)take(FF * 4);
    float* bnsq  = (float*)take(FF * 4);
    (void)ws_size; (void)in_sizes; (void)n_in; (void)out_size;

    // K0: pack weights to f16, transposed ([w][n][k])
    pack_weights_kernel<<<(4 * FF * FF + 255) / 256, 256, 0, stream>>>(Wq, Wk, Wv, Wsk, wpack);

    // K1: init scratch (covers NN*FF elements)
    init_kernel<<<(NN * FF + 255) / 256, 256, 0, stream>>>(amax, denom, agg, bnsum, bnsq);

    // K2: fused projections (WMMA). grid: 3125 row-tiles x 8 col-tiles, 1 wave/block.
    gemm_qkvs_kernel<<<dim3(NN / 16, FF / 16), 32, 0, stream>>>(
        x, wpack, bq, bk, bv, bsk, q, k2, v2, skipb);

    // K3: attention logits + segment max (wave per edge, 8 waves/block)
    edge_logits_kernel<<<(EE + 7) / 8, 256, 0, stream>>>(ei, ea, We, q, k2, alpha, amax);

    // K4: exp + segment sum (thread per edge-head)
    edge_expsum_kernel<<<(EE * HH + 255) / 256, 256, 0, stream>>>(ei, alpha, amax, denom);

    // K5: weighted messages, scatter-add (wave per edge)
    edge_message_kernel<<<(EE + 7) / 8, 256, 0, stream>>>(ei, ea, We, v2, alpha, denom, agg);

    // K6: skip connection + BN statistics
    skip_stats_kernel<<<(NN + 63) / 64, 128, 0, stream>>>(skipb, agg, bnsum, bnsq);

    // K7: BatchNorm + LeakyReLU -> d_out
    finalize_kernel<<<(NN * FF + 255) / 256, 256, 0, stream>>>(agg, bnsum, bnsq, gma, bta, out);
}
